// RNN_LSTM_12395275616626
// MI455X (gfx1250) — matrix operand order repaired and checked by reading, not compiled, not run on hardware
//
#include <hip/hip_runtime.h>
#include <stdint.h>

// ---------------------------------------------------------------------------
// 2-layer LSTM (T=32768, I=12, H=256) + linear head, for MI455X (gfx1250).
//
// Pipeline (all on `stream`):
//   0. init_sync        : zero the two grid-barrier counters
//   1. prep_weights     : W_hh0/W_hh1 -> bf16 ; W_ih1 -> bf16 pair-packed B-layout
//   2. xp0              : xp = x @ W_ih0^T + (b_ih0+b_hh0)        (K=12, scalar)
//   3. lstm_scan (L0)   : persistent 4-WG kernel, W_hh in LDS via TDM
//   4. h1 -> bf16 pairs : A-matrix for the WMMA GEMM
//   5. xp1 (WMMA bf16)  : xp = h1 @ W_ih1^T + (b_ih1+b_hh1)       (32768x1024x256)
//   6. lstm_scan (L1)
//   7. fc               : out = h2 @ fc_W^T + fc_b                (T x 3)
// ---------------------------------------------------------------------------

#define SEQ_T 32768
#define NWG   4            // workgroups in the persistent scan grid
#define ROW_BYTES 520u     // 256 bf16 (512B) + 2-dword pad -> conflict-free wave32 b64 reads

typedef __bf16 v16bf __attribute__((ext_vector_type(16)));
typedef float  v8f   __attribute__((ext_vector_type(8)));
typedef unsigned int v4u __attribute__((ext_vector_type(4)));
typedef int          v8i __attribute__((ext_vector_type(8)));
typedef int          v4i __attribute__((ext_vector_type(4)));

// ----------------------------- helpers -------------------------------------
__device__ __forceinline__ unsigned short f2bf(float f) {
  unsigned u = __float_as_uint(f);
  unsigned r = u + 0x7FFFu + ((u >> 16) & 1u);   // round-to-nearest-even
  return (unsigned short)(r >> 16);
}
__device__ __forceinline__ float bflo(unsigned u) { return __uint_as_float(u << 16); }
__device__ __forceinline__ float bfhi(unsigned u) { return __uint_as_float(u & 0xFFFF0000u); }
__device__ __forceinline__ float sigf(float x)    { return 1.0f / (1.0f + __expf(-x)); }

// Tensor Data Mover: toolchain-portable wrapper (5-arg ROCm7.2 vs 6-arg therock)
#if defined(__has_builtin)
#if __has_builtin(__builtin_amdgcn_tensor_load_to_lds)
#define HAVE_TDM 1
#endif
#endif
#if defined(HAVE_TDM)
#if __has_include(<hip/amd_detail/amd_gfx1250_TDM.h>)
#define TDM_LOAD(g0, g1, g2, g3) \
  __builtin_amdgcn_tensor_load_to_lds((g0), (g1), (g2), (g3), (v8i){0,0,0,0,0,0,0,0}, 0)
#else
#define TDM_LOAD(g0, g1, g2, g3) \
  __builtin_amdgcn_tensor_load_to_lds((g0), (g1), (g2), (g3), 0)
#endif
#endif

// ----------------------------- kernels -------------------------------------

__global__ void init_sync_kernel(unsigned* cnt) {
  if (threadIdx.x < 2) cnt[threadIdx.x] = 0u;
}

// W_hh{0,1} (1024x256 f32) -> bf16 row-major; W_ih1 (1024x256 f32) -> pair-packed
// B layout: Bp[k2*1024 + n] = pack(W_ih1[n][2k2], W_ih1[n][2k2+1])
__global__ void prep_weights_kernel(const float* __restrict__ Whh0,
                                    const float* __restrict__ Whh1,
                                    const float* __restrict__ Wih1,
                                    unsigned short* __restrict__ whh0b,
                                    unsigned short* __restrict__ whh1b,
                                    unsigned* __restrict__ Bp) {
  int i = blockIdx.x * 256 + threadIdx.x;        // 0 .. 1024*256-1
  if (i < 1024 * 256) {
    whh0b[i] = f2bf(Whh0[i]);
    whh1b[i] = f2bf(Whh1[i]);
  }
  if (i < 128 * 1024) {
    int k2 = i >> 10, n = i & 1023;
    unsigned lo = f2bf(Wih1[n * 256 + 2 * k2]);
    unsigned hi = f2bf(Wih1[n * 256 + 2 * k2 + 1]);
    Bp[i] = lo | (hi << 16);
  }
}

// xp[t, row] = b_ih0[row] + b_hh0[row] + sum_k x[t,k] * W_ih0[row,k]  (K = 12)
__global__ void xp0_kernel(const float* __restrict__ x, const float* __restrict__ Wih0,
                           const float* __restrict__ bi, const float* __restrict__ bh,
                           float* __restrict__ xp) {
  size_t i = (size_t)blockIdx.x * 256 + threadIdx.x;   // T*1024 threads
  int row = (int)(i & 1023);
  size_t t = i >> 10;
  const float* xr = x + t * 12;
  const float* wr = Wih0 + (size_t)row * 12;
  float acc = bi[row] + bh[row];
#pragma unroll
  for (int k = 0; k < 12; ++k) acc = fmaf(xr[k], wr[k], acc);
  xp[i] = acc;
}

// Persistent 4-WG LSTM scan. WG `wg` owns h elements [64*wg, 64*wg+64) and the
// 4 gate rows per element. Weights live in LDS (bf16, padded rows) for all T
// steps; loaded once via the Tensor Data Mover. One grid sync per step with a
// double-buffered h exchange through L2.
__launch_bounds__(256, 1)
__global__ void lstm_scan_kernel(const float* __restrict__ xp,
                                 const unsigned short* __restrict__ whhB,
                                 float* __restrict__ hout,
                                 float* hx, unsigned* cnt, int Tlen) {
  __shared__ unsigned char wsm[256 * ROW_BYTES];   // 130 KB weight slab
  __shared__ float h_cur[256];
  __shared__ float gbuf[256];

  const int tid = threadIdx.x;
  const int wg  = blockIdx.x;           // 0..3
  const int g   = tid >> 6;             // gate: 0=i 1=f 2=g 3=o
  const int j   = tid & 63;             // element within slice

  // ---- stage W_hh rows into LDS (once) -------------------------------------
#if defined(HAVE_TDM)
  if (tid < 32) {                       // wave 0 issues 4 TDM tiles (64 rows each)
    unsigned ldsbase = (unsigned)(size_t)(void*)wsm;
#pragma unroll
    for (int gg = 0; gg < 4; ++gg) {
      unsigned long long ga =
          (unsigned long long)(size_t)(whhB + ((size_t)(gg * 256 + wg * 64)) * 256);
      unsigned lds_off = ldsbase + (unsigned)gg * 64u * ROW_BYTES;
      // D# group0: count=1 | lds_addr | global_addr[56:0] | type=2
      v4u g0 = { 1u, lds_off, (unsigned)ga,
                 ((unsigned)((ga >> 32) & 0x1FFFFFFu)) | 0x80000000u };
      // D# group1: data_size=2B, pad_enable, pad_interval=128 DW, pad_amount=2 DW,
      //            tensor_dim0=256, tensor_dim1=64, tile_dim0=256, tile_dim1=64,
      //            tensor_dim0_stride=256
      v8i g1 = { (int)0x03910000,        // ds=1<<16 | pad_en=1<<20 | intv6<<22 | amt1<<25
                 (int)(256 << 16),       // tensor_dim0[15:0] @ bits 63:48
                 (int)(64  << 16),       // tensor_dim1[15:0] @ bits 95:80... (111:80)
                 (int)(256 << 16),       // tile_dim0 @ bits 127:112
                 64,                     // tile_dim1 @ bits 143:128
                 256,                    // tensor_dim0_stride[31:0]
                 0, 0 };
      v4i g2 = {0, 0, 0, 0};
      v4i g3 = {0, 0, 0, 0};
      TDM_LOAD(g0, g1, g2, g3);
    }
    __builtin_amdgcn_s_wait_tensorcnt(0);
  }
#else
  // Fallback: cooperative copy with the same padded layout.
  for (int r = tid >> 3; r < 256; r += 32) {      // 8 threads per row
    int grow = (r >> 6) * 256 + wg * 64 + (r & 63);
    const unsigned* src = (const unsigned*)(whhB + (size_t)grow * 256);
    unsigned* dst = (unsigned*)(wsm + (size_t)r * ROW_BYTES);
    for (int q = tid & 7; q < 128; q += 8) dst[q] = src[q];
  }
#endif
  h_cur[tid] = 0.0f;
  __syncthreads();

  float c = 0.0f;
  const int   row    = g * 256 + wg * 64 + j;     // gate row in [0,1024)
  const uint2* wrow  = (const uint2*)(wsm + (size_t)tid * ROW_BYTES);
  const float4* h4   = (const float4*)h_cur;

  for (int t = 0; t < Tlen; ++t) {
    float xv = xp[(size_t)t * 1024 + row];
    if (t + 1 < Tlen) __builtin_prefetch(&xp[(size_t)(t + 1) * 1024 + row], 0, 0);

    // gate pre-activation: 256-deep dot, bf16 weights from LDS, f32 accumulate
    float acc = xv;
#pragma unroll 8
    for (int q = 0; q < 64; ++q) {
      uint2  u  = wrow[q];
      float4 hv = h4[q];
      acc = fmaf(bflo(u.x), hv.x, acc);
      acc = fmaf(bfhi(u.x), hv.y, acc);
      acc = fmaf(bflo(u.y), hv.z, acc);
      acc = fmaf(bfhi(u.y), hv.w, acc);
    }
    gbuf[tid] = acc;
    __syncthreads();

    if (tid < 64) {                     // combine i,f,g,o -> c,h for owned slice
      float iv = sigf(gbuf[tid]);
      float fv = sigf(gbuf[64 + tid]);
      float gv = tanhf(gbuf[128 + tid]);
      float ov = sigf(gbuf[192 + tid]);
      c = fv * c + iv * gv;
      float hval = ov * tanhf(c);
      hout[(size_t)t * 256 + wg * 64 + tid] = hval;
      __hip_atomic_store(&hx[(t & 1) * 256 + wg * 64 + tid], hval,
                         __ATOMIC_RELEASE, __HIP_MEMORY_SCOPE_AGENT);
    }
    __syncthreads();

    if (tid == 0) {                     // 4-WG grid barrier (monotone counter)
      atomicAdd(cnt, 1u);
      unsigned target = (unsigned)(t + 1) * (unsigned)NWG;
      while (__hip_atomic_load(cnt, __ATOMIC_ACQUIRE, __HIP_MEMORY_SCOPE_AGENT) < target)
        __builtin_amdgcn_s_sleep(1);
    }
    __syncthreads();

    h_cur[tid] = __hip_atomic_load(&hx[(t & 1) * 256 + tid],
                                   __ATOMIC_RELAXED, __HIP_MEMORY_SCOPE_AGENT);
    __syncthreads();
  }
}

// h1 (T x 256 f32) -> pair-packed bf16 (A matrix for WMMA)
__global__ void h1cvt_kernel(const float* __restrict__ h1, unsigned* __restrict__ h1b) {
  size_t i = (size_t)blockIdx.x * 256 + threadIdx.x;    // T*128 threads
  float a = h1[2 * i], b = h1[2 * i + 1];
  h1b[i] = (unsigned)f2bf(a) | ((unsigned)f2bf(b) << 16);
}

// xp = h1 @ W_ih1^T + (b_ih1 + b_hh1) via v_wmma_f32_16x16x32_bf16.
// Per-wave 16x16 tile; A/B operand layouts follow the CDNA5 16-bit striping.
__launch_bounds__(256)
__global__ void xp1_wmma_kernel(const unsigned* __restrict__ h1b,   // T x 128 dword pairs
                                const unsigned* __restrict__ Bp,    // 128 x 1024 dword pairs
                                const float* __restrict__ b_ih1,
                                const float* __restrict__ b_hh1,
                                float* __restrict__ xp) {
  const int lane  = threadIdx.x & 31;
  const int wave  = threadIdx.x >> 5;
  const int mBase = blockIdx.x * 16;
  const int nBase = (blockIdx.y * 8 + wave) * 16;
  const int hi    = lane >> 4;          // half-wave select
  const int lm    = lane & 15;
  const int r     = mBase + lm;         // A row (time step)
  const int n     = nBase + lm;         // B col (gate row)
  const float bias = b_ih1[n] + b_hh1[n];

  union { unsigned u[8]; v16bf v; } A, B;
  v8f acc = {};

  for (int k = 0; k < 256; k += 32) {
#pragma unroll
    for (int jj = 0; jj < 8; ++jj) {
      // A 16x32 bf16: VGPR jj holds K = kk,kk+1 per documented striping
      int kk = k + ((jj >> 2) << 4) + hi * 8 + ((jj & 3) << 1);
      A.u[jj] = h1b[(size_t)r * 128 + (kk >> 1)];
      // B 32x16 bf16: VGPR jj holds same K pair as A (table order)
      int kp = (k >> 1) + ((jj >> 2) << 3) + hi * 4 + (jj & 3);
      B.u[jj] = Bp[(size_t)kp * 1024 + n];
    }
    acc = __builtin_amdgcn_wmma_f32_16x16x32_bf16(
        false, A.v, false, B.v, (short)0, acc, false, false);
  }
#pragma unroll
  for (int v = 0; v < 8; ++v) {
    int M = mBase + hi * 8 + v;         // C/D layout: vgpr v -> M = v (+8 for hi lanes)
    xp[(size_t)M * 1024 + n] = acc[v] + bias;
  }
}

// out[t] = h2[t] @ fc_W^T + fc_b   (C = 3)
__global__ void fc_kernel(const float* __restrict__ h2, const float* __restrict__ fcW,
                          const float* __restrict__ fcb, float* __restrict__ out, int Tlen) {
  __shared__ float w[3 * 256];
  for (int i = threadIdx.x; i < 3 * 256; i += 256) w[i] = fcW[i];
  __syncthreads();
  int t = blockIdx.x * 256 + threadIdx.x;
  if (t >= Tlen) return;
  const float4* hr = (const float4*)(h2 + (size_t)t * 256);
  float a0 = fcb[0], a1 = fcb[1], a2 = fcb[2];
#pragma unroll 4
  for (int q = 0; q < 64; ++q) {
    float4 hv = hr[q];
    int k = q * 4;
    a0 = fmaf(hv.x, w[k], a0);       a1 = fmaf(hv.x, w[256 + k], a1);       a2 = fmaf(hv.x, w[512 + k], a2);
    a0 = fmaf(hv.y, w[k + 1], a0);   a1 = fmaf(hv.y, w[256 + k + 1], a1);   a2 = fmaf(hv.y, w[512 + k + 1], a2);
    a0 = fmaf(hv.z, w[k + 2], a0);   a1 = fmaf(hv.z, w[256 + k + 2], a1);   a2 = fmaf(hv.z, w[512 + k + 2], a2);
    a0 = fmaf(hv.w, w[k + 3], a0);   a1 = fmaf(hv.w, w[256 + k + 3], a1);   a2 = fmaf(hv.w, w[512 + k + 3], a2);
  }
  out[(size_t)t * 3 + 0] = a0;
  out[(size_t)t * 3 + 1] = a1;
  out[(size_t)t * 3 + 2] = a2;
}

// ----------------------------- host launch ----------------------------------
extern "C" void kernel_launch(void* const* d_in, const int* in_sizes, int n_in,
                              void* d_out, int out_size, void* d_ws, size_t ws_size,
                              hipStream_t stream) {
  (void)in_sizes; (void)n_in; (void)out_size; (void)ws_size;
  const float* x     = (const float*)d_in[0];
  const float* Wih0  = (const float*)d_in[1];
  const float* Whh0  = (const float*)d_in[2];
  const float* bih0  = (const float*)d_in[3];
  const float* bhh0  = (const float*)d_in[4];
  const float* Wih1  = (const float*)d_in[5];
  const float* Whh1  = (const float*)d_in[6];
  const float* bih1  = (const float*)d_in[7];
  const float* bhh1  = (const float*)d_in[8];
  const float* fcW   = (const float*)d_in[9];
  const float* fcb   = (const float*)d_in[10];
  float* out = (float*)d_out;

  char* w = (char*)d_ws;                                // workspace carve-up
  const size_t T = SEQ_T;
  float*          xp    = (float*)(w);                                  // T*1024 f32 (shared by L0/L1)
  float*          h1    = (float*)(w + 134217728);                      // T*256 f32
  float*          h2    = (float*)(w + 167772160);                      // T*256 f32
  unsigned*       h1b   = (unsigned*)(w + 201326592);                   // T*128 dwords
  unsigned short* whh0b = (unsigned short*)(w + 218103808);             // 1024*256 bf16
  unsigned short* whh1b = (unsigned short*)(w + 218628096);
  unsigned*       Bp    = (unsigned*)(w + 219152384);                   // 128*1024 dwords
  float*          hx    = (float*)(w + 219676672);                      // 2*256 f32 exchange
  unsigned*       cnt   = (unsigned*)(w + 219678720);                   // 2 barrier counters

  init_sync_kernel<<<1, 32, 0, stream>>>(cnt);
  prep_weights_kernel<<<1024, 256, 0, stream>>>(Whh0, Whh1, Wih1, whh0b, whh1b, Bp);
  xp0_kernel<<<(unsigned)(T * 1024 / 256), 256, 0, stream>>>(x, Wih0, bih0, bhh0, xp);

  lstm_scan_kernel<<<NWG, 256, 0, stream>>>(xp, whh0b, h1, hx, cnt + 0, (int)T);

  h1cvt_kernel<<<(unsigned)(T * 128 / 256), 256, 0, stream>>>(h1, h1b);
  dim3 ggrid((unsigned)(T / 16), 1024 / (16 * 8), 1);
  xp1_wmma_kernel<<<ggrid, 256, 0, stream>>>(h1b, Bp, bih1, bhh1, xp);

  lstm_scan_kernel<<<NWG, 256, 0, stream>>>(xp, whh1b, h2, hx, cnt + 1, (int)T);

  fc_kernel<<<(unsigned)(T / 256), 256, 0, stream>>>(h2, fcW, fcb, out, (int)T);
}
